// NonLocalBlock_10960756539394
// MI455X (gfx1250) — compile-verified, matrix operand order checked by reading, not run on hardware
//
#include <hip/hip_runtime.h>
#include <hip/hip_bf16.h>

#define KN 4
#define KC 256
#define KD 128
#define KS 4096   // 64*64

typedef __attribute__((ext_vector_type(8)))  __bf16 bf16x8;
typedef __attribute__((ext_vector_type(16))) __bf16 bf16x16;
typedef __attribute__((ext_vector_type(8)))  float  floatx8;

__device__ __forceinline__ floatx8 fzero() {
  floatx8 z;
#pragma unroll
  for (int i = 0; i < 8; ++i) z[i] = 0.f;
  return z;
}

// Load a 16x32 bf16 WMMA operand from a row-major matrix (leading dim `ld`
// elements). Works for the A operand (rows = M) and for the B operand when
// the source is stored transposed (rows = N, i.e. column-major B).
// Lane L holds row (row0 + L%16); VGPRs 0..3 <- K = k0 + (L/16)*8 + 0..7,
// VGPRs 4..7 <- K = k0 + 16 + (L/16)*8 + 0..7  (matches ISA 16-bit A layout).
__device__ __forceinline__ bf16x16 load_tile(const __bf16* p, int row0, int ld,
                                             int k0, int lane) {
  const int m = lane & 15, h = lane >> 4;
  const __bf16* base = p + (size_t)(row0 + m) * ld + k0 + h * 8;
  bf16x8 lo = *(const bf16x8*)base;
  bf16x8 hi = *(const bf16x8*)(base + 16);
  bf16x16 r;
#pragma unroll
  for (int i = 0; i < 8; ++i) { r[i] = lo[i]; r[8 + i] = hi[i]; }
  return r;
}

__device__ __forceinline__ floatx8 wmma_bf16(bf16x16 a, bf16x16 b, floatx8 c) {
  return __builtin_amdgcn_wmma_f32_16x16x32_bf16(false, a, false, b,
                                                 (short)0, c, false, false);
}

// gfx1250 async copy: global -> LDS, no VGPR data path, tracked by ASYNCcnt.
// VDST operand is the raw LDS byte offset (wave LDS_BASE relative).
__device__ __forceinline__ void async_b128(unsigned ldsoff, const void* gsrc) {
  asm volatile("global_load_async_to_lds_b128 %0, %1, off"
               :: "v"(ldsoff), "v"(gsrc) : "memory");
}
__device__ __forceinline__ void wait_async0() {
  asm volatile("s_wait_asynccnt 0" ::: "memory");
}

// ---------------------------------------------------------------------------
// Kernel 1: x (N,C,S) f32 -> xT (N,S,C) bf16, LDS-tiled transpose.
// ---------------------------------------------------------------------------
__global__ __launch_bounds__(256) void k_transpose(const float* __restrict__ x,
                                                   __bf16* __restrict__ xT) {
  __shared__ __bf16 tile[32][33];
  const int n = blockIdx.z;
  const int s0 = blockIdx.x * 32, c0 = blockIdx.y * 32;
  const float* xp = x + (size_t)n * KC * KS;
  __bf16* xtp = xT + (size_t)n * KS * KC;
#pragma unroll
  for (int r = threadIdx.y; r < 32; r += 8)
    tile[r][threadIdx.x] = (__bf16)xp[(size_t)(c0 + r) * KS + s0 + threadIdx.x];
  __syncthreads();
#pragma unroll
  for (int r = threadIdx.y; r < 32; r += 8)
    xtp[(size_t)(s0 + r) * KC + c0 + threadIdx.x] = tile[threadIdx.x][r];
}

// ---------------------------------------------------------------------------
// Kernel 2: convert the four weight matrices (each D*C = 32768 elems) to bf16.
// ---------------------------------------------------------------------------
__global__ __launch_bounds__(256) void k_cvt_w(
    const float* __restrict__ wg, const float* __restrict__ wt,
    const float* __restrict__ wp, const float* __restrict__ wz,
    __bf16* __restrict__ og, __bf16* __restrict__ ot,
    __bf16* __restrict__ op, __bf16* __restrict__ oz) {
  const int i = blockIdx.x * 256 + threadIdx.x;
  og[i] = (__bf16)wg[i];
  ot[i] = (__bf16)wt[i];
  op[i] = (__bf16)wp[i];
  oz[i] = (__bf16)wz[i];
}

// ---------------------------------------------------------------------------
// Kernel 3: thT/phT (N,S,D) bf16.  thT[s,d] = sum_c xT[s,c]*Wt[d,c] + bt[d].
// ---------------------------------------------------------------------------
__global__ __launch_bounds__(128) void k_proj_qk(
    const __bf16* __restrict__ xT, const __bf16* __restrict__ wt,
    const __bf16* __restrict__ wp, const float* __restrict__ bt,
    const float* __restrict__ bp, __bf16* __restrict__ thT,
    __bf16* __restrict__ phT) {
  const int lane = threadIdx.x & 31, wave = threadIdx.x >> 5;
  const int n = blockIdx.y;
  const int s0 = (blockIdx.x * 4 + wave) * 16;
  const int nl = lane & 15, h = lane >> 4;
  const __bf16* an = xT + (size_t)n * KS * KC;

  bf16x16 a[8];
#pragma unroll
  for (int kc = 0; kc < 8; ++kc) a[kc] = load_tile(an, s0, KC, kc * 32, lane);

  __bf16* tn = thT + (size_t)n * KS * KD;
  __bf16* pn = phT + (size_t)n * KS * KD;
#pragma unroll
  for (int dt = 0; dt < 8; ++dt) {
    floatx8 ct = fzero(), cp = fzero();
#pragma unroll
    for (int kc = 0; kc < 8; ++kc) {
      ct = wmma_bf16(a[kc], load_tile(wt, dt * 16, KC, kc * 32, lane), ct);
      cp = wmma_bf16(a[kc], load_tile(wp, dt * 16, KC, kc * 32, lane), cp);
    }
    const int d = dt * 16 + nl;
    const float biast = bt[d], biasp = bp[d];
#pragma unroll
    for (int r = 0; r < 8; ++r) {
      const int s = s0 + r + 8 * h;
      tn[(size_t)s * KD + d] = (__bf16)(ct[r] + biast);
      pn[(size_t)s * KD + d] = (__bf16)(cp[r] + biasp);
    }
  }
}

// ---------------------------------------------------------------------------
// Kernel 4: g (N,D,S) bf16.  g[d,s] = sum_c Wg[d,c]*x[c,s] + bg[d].
// ---------------------------------------------------------------------------
__global__ __launch_bounds__(128) void k_proj_g(
    const __bf16* __restrict__ xT, const __bf16* __restrict__ wg,
    const float* __restrict__ bg, __bf16* __restrict__ g) {
  const int lane = threadIdx.x & 31, wave = threadIdx.x >> 5;
  const int n = blockIdx.y;
  const int s0 = (blockIdx.x * 4 + wave) * 16;
  const int nl = lane & 15, h = lane >> 4;

  bf16x16 b[8];
  const __bf16* an = xT + (size_t)n * KS * KC;
#pragma unroll
  for (int kc = 0; kc < 8; ++kc) b[kc] = load_tile(an, s0, KC, kc * 32, lane);

  __bf16* gn = g + (size_t)n * KD * KS;
#pragma unroll
  for (int mt = 0; mt < 8; ++mt) {
    floatx8 c = fzero();
#pragma unroll
    for (int kc = 0; kc < 8; ++kc)
      c = wmma_bf16(load_tile(wg, mt * 16, KC, kc * 32, lane), b[kc], c);
#pragma unroll
    for (int r = 0; r < 8; ++r) {
      const int d = mt * 16 + r + 8 * h;
      gn[(size_t)d * KS + s0 + nl] = (__bf16)(c[r] + bg[d]);
    }
  }
}

// ---------------------------------------------------------------------------
// Kernel 5: fused flash-attention + output projection + bias + residual.
// 4 waves/WG share double-buffered LDS stages of the phT (32x128) and g
// (128x32) t-block tiles, filled with gfx1250 async global->LDS copies
// (ASYNCcnt) while WMMA consumes the previous block.  Scores are computed
// transposed so the f32 accumulator layout doubles as the bf16 B-operand
// layout of the next GEMM (register-only softmax->P packing); y accumulators
// chain the same way into the Wz projection.
// ---------------------------------------------------------------------------
#define PH_BYTES (32 * KD * 2)     // 8 KB
#define G_BYTES  (KD * 32 * 2)     // 8 KB
#define BUF_BYTES (PH_BYTES + G_BYTES)

__global__ __launch_bounds__(128) void k_attn_out(
    const __bf16* __restrict__ thT, const __bf16* __restrict__ phT,
    const __bf16* __restrict__ g, const __bf16* __restrict__ wz,
    const float* __restrict__ bz, const float* __restrict__ x,
    float* __restrict__ out) {
  extern __shared__ char smem[];   // 2 * (8KB phT stage + 8KB g stage)
  const int tid = threadIdx.x;
  const int lane = tid & 31, wave = tid >> 5;
  const int n = blockIdx.y;
  const int s0 = (blockIdx.x * 4 + wave) * 16;
  const int nl = lane & 15, h = lane >> 4;
  const float LOG2E = 1.4426950408889634f;

  const __bf16* thn = thT + (size_t)n * KS * KD;
  const __bf16* phn = phT + (size_t)n * KS * KD;
  const __bf16* gn = g + (size_t)n * KD * KS;

  // Issue this thread's share of async copies for t-block tb into buffer buf.
  auto stage = [&](int buf, int tb) {
    const unsigned base = (unsigned)buf * BUF_BYTES;
#pragma unroll
    for (int i = 0; i < 4; ++i) {          // phT tile: 512 x 16B chunks
      const int c = tid + i * 128;
      const int row = c >> 4, col = c & 15;          // [32][128] bf16
      async_b128(base + (unsigned)c * 16,
                 phn + (size_t)(tb + row) * KD + col * 8);
    }
#pragma unroll
    for (int i = 0; i < 4; ++i) {          // g tile: 512 x 16B chunks
      const int c = tid + i * 128;
      const int row = c >> 2, col = c & 3;           // [128][32] bf16
      async_b128(base + PH_BYTES + (unsigned)c * 16,
                 gn + (size_t)row * KS + tb + col * 8);
    }
  };

  // B operand of score GEMM: columns s of th (rows of thT). Fixed per wave.
  bf16x16 bth[4];
#pragma unroll
  for (int kk = 0; kk < 4; ++kk) bth[kk] = load_tile(thn, s0, KD, kk * 32, lane);

  floatx8 acc[8];
#pragma unroll
  for (int m = 0; m < 8; ++m) acc[m] = fzero();
  float mrow = -1e30f, lrow = 0.f;

  stage(0, 0);
  wait_async0();
  __syncthreads();

  int buf = 0;
  for (int tb = 0; tb < KS; tb += 32) {
    if (tb + 32 < KS) stage(buf ^ 1, tb + 32);

    const __bf16* lph = (const __bf16*)(smem + buf * BUF_BYTES);       // [32][128]
    const __bf16* lg = (const __bf16*)(smem + buf * BUF_BYTES + PH_BYTES); // [128][32]

    // fT[t, s] = sum_d phT[t,d] * thT[s,d] ; two 16-row t tiles, K = D = 128.
    floatx8 sc0 = fzero(), sc1 = fzero();
#pragma unroll
    for (int kk = 0; kk < 4; ++kk)
      sc0 = wmma_bf16(load_tile(lph, 0, KD, kk * 32, lane), bth[kk], sc0);
#pragma unroll
    for (int kk = 0; kk < 4; ++kk)
      sc1 = wmma_bf16(load_tile(lph, 16, KD, kk * 32, lane), bth[kk], sc1);

    // Online softmax: per s = lane%16; combine the two half-wave t-ranges.
    float bm = -1e30f;
#pragma unroll
    for (int r = 0; r < 8; ++r) bm = fmaxf(bm, fmaxf(sc0[r], sc1[r]));
    bm = fmaxf(bm, __shfl_xor(bm, 16, 32));
    const float mnew = fmaxf(mrow, bm);
    const float corr = exp2f((mrow - mnew) * LOG2E);
#pragma unroll
    for (int m = 0; m < 8; ++m)
#pragma unroll
      for (int r = 0; r < 8; ++r) acc[m][r] *= corr;
    float bs = 0.f;
#pragma unroll
    for (int r = 0; r < 8; ++r) {
      const float e0 = exp2f((sc0[r] - mnew) * LOG2E);
      const float e1 = exp2f((sc1[r] - mnew) * LOG2E);
      sc0[r] = e0; sc1[r] = e1; bs += e0 + e1;
    }
    bs += __shfl_xor(bs, 16, 32);
    lrow = lrow * corr + bs;
    mrow = mnew;

    // Pack P^T (D-layout) straight into the B operand (K=t chunk of 32).
    bf16x16 pb;
#pragma unroll
    for (int v = 0; v < 8; ++v) {
      const int j = (v & 3) * 2;
      if (v < 4) { pb[2 * v] = (__bf16)sc0[j]; pb[2 * v + 1] = (__bf16)sc0[j + 1]; }
      else       { pb[2 * v] = (__bf16)sc1[j]; pb[2 * v + 1] = (__bf16)sc1[j + 1]; }
    }
    // y[d, s] += g[d, t] * P[t, s]  ; g tile from LDS ([128][32], k0 = 0).
#pragma unroll
    for (int m = 0; m < 8; ++m)
      acc[m] = wmma_bf16(load_tile(lg, m * 16, 32, 0, lane), pb, acc[m]);

    wait_async0();   // next block's copies landed (this wave's share)
    __syncthreads(); // all waves done reading buf & writing buf^1
    buf ^= 1;
  }

  const float rinv = 1.f / lrow;
#pragma unroll
  for (int m = 0; m < 8; ++m)
#pragma unroll
    for (int r = 0; r < 8; ++r) acc[m][r] *= rinv;

  // Pack y (D-layout, M=d N=s) into B operands for out = Wz * y (K = D = 128).
  bf16x16 by[4];
#pragma unroll
  for (int kk = 0; kk < 4; ++kk)
#pragma unroll
    for (int v = 0; v < 8; ++v) {
      const int tile = 2 * kk + (v >> 2), j = (v & 3) * 2;
      by[kk][2 * v] = (__bf16)acc[tile][j];
      by[kk][2 * v + 1] = (__bf16)acc[tile][j + 1];
    }

  const float* xn = x + (size_t)n * KC * KS;
  float* on = out + (size_t)n * KC * KS;
#pragma unroll
  for (int ct = 0; ct < 16; ++ct) {
    floatx8 o = fzero();
#pragma unroll
    for (int kk = 0; kk < 4; ++kk)
      o = wmma_bf16(load_tile(wz, ct * 16, KD, kk * 32, lane), by[kk], o);
#pragma unroll
    for (int r = 0; r < 8; ++r) {
      const int c = ct * 16 + r + 8 * h;
      const size_t idx = (size_t)c * KS + s0 + nl;
      on[idx] = o[r] + bz[c] + xn[idx];
    }
  }
}

// ---------------------------------------------------------------------------
extern "C" void kernel_launch(void* const* d_in, const int* in_sizes, int n_in,
                              void* d_out, int out_size, void* d_ws,
                              size_t ws_size, hipStream_t stream) {
  const float* x  = (const float*)d_in[0];
  const float* Wg = (const float*)d_in[1];
  const float* bg = (const float*)d_in[2];
  const float* Wt = (const float*)d_in[3];
  const float* bt = (const float*)d_in[4];
  const float* Wp = (const float*)d_in[5];
  const float* bp = (const float*)d_in[6];
  const float* Wz = (const float*)d_in[7];
  const float* bz = (const float*)d_in[8];
  float* out = (float*)d_out;

  char* ws = (char*)d_ws;
  size_t off = 0;
  auto alloc = [&](size_t bytes) -> void* {
    off = (off + 255) & ~(size_t)255;
    void* p = ws + off;
    off += bytes;
    return p;
  };
  __bf16* xT  = (__bf16*)alloc((size_t)KN * KS * KC * 2);  // 8 MB
  __bf16* wgb = (__bf16*)alloc((size_t)KD * KC * 2);
  __bf16* wtb = (__bf16*)alloc((size_t)KD * KC * 2);
  __bf16* wpb = (__bf16*)alloc((size_t)KD * KC * 2);
  __bf16* wzb = (__bf16*)alloc((size_t)KC * KD * 2);
  __bf16* thT = (__bf16*)alloc((size_t)KN * KS * KD * 2);  // 4 MB
  __bf16* phT = (__bf16*)alloc((size_t)KN * KS * KD * 2);  // 4 MB
  __bf16* gb  = (__bf16*)alloc((size_t)KN * KD * KS * 2);  // 4 MB
  (void)ws_size; (void)in_sizes; (void)n_in; (void)out_size;

  k_transpose<<<dim3(KS / 32, KC / 32, KN), dim3(32, 8), 0, stream>>>(x, xT);
  k_cvt_w<<<dim3(KD * KC / 256), 256, 0, stream>>>(Wg, Wt, Wp, Wz,
                                                   wgb, wtb, wpb, wzb);
  k_proj_qk<<<dim3(KS / 64, KN), 128, 0, stream>>>(xT, wtb, wpb, bt, bp,
                                                   thT, phT);
  k_proj_g<<<dim3(KS / 64, KN), 128, 0, stream>>>(xT, wgb, bg, gb);
  k_attn_out<<<dim3(KS / 64, KN), 128, 2 * BUF_BYTES, stream>>>(
      thT, phT, gb, wzb, bz, x, out);
}
